// RGCNLayer_26757646254162
// MI455X (gfx1250) — matrix operand order repaired and checked by reading, not compiled, not run on hardware
//
#include <hip/hip_runtime.h>
#include <hip/hip_bf16.h>

// ---------------------------------------------------------------------------
// RGCN layer (basis decomposition) for MI455X / gfx1250.
// N=50000, E=400000, R=1344, B=4, D=128.
//   hb  = node_feats @ basis          (WMMA bf16 GEMM, M=50000 K=128 N=512)
//   msg = (comp[etype] . hb[src]) * norm ; agg = scatter_add(msg, dst)
//   h   = relu(agg + bias) + x ; out = relu(BN(h))
// hb (102.4MB) stays resident in the 192MB L2, so the 800MB edge gather is an
// L2-bandwidth problem, not an HBM one. B is pre-swizzled to WMMA fragment
// order so every fragment load is a contiguous 128-bit vector load.
// ---------------------------------------------------------------------------

typedef __attribute__((ext_vector_type(16))) __bf16         v16bf;
typedef __attribute__((ext_vector_type(8)))  float          v8f;
typedef __attribute__((ext_vector_type(8)))  unsigned short v8us;
typedef __attribute__((ext_vector_type(16))) unsigned short v16us;

#define N_NODES 50000
#define N_EDGES 400000
#define N_RELS  1344
#define N_BASES 4
#define DIM     128
#define NCOLS   (N_BASES * DIM)   // 512
#define BN_EPS  1e-5f

__device__ __forceinline__ unsigned short f2bf_bits(float f) {
    unsigned int u = __builtin_bit_cast(unsigned int, f);
    u += 0x7fffu + ((u >> 16) & 1u);        // round-to-nearest-even
    return (unsigned short)(u >> 16);
}

// --- bf16 conversion of node features (row-major, K-contiguous) --------------
__global__ __launch_bounds__(256)
void k_conv_a(const float* __restrict__ x, unsigned short* __restrict__ abf, int total) {
    int i = blockIdx.x * 256 + threadIdx.x;
    if (i < total) abf[i] = f2bf_bits(x[i]);
}

// --- bf16 conversion + WMMA-fragment swizzle of basis ------------------------
// Logical B[k][col], col = b*128 + o, k = input dim.
// Fragment order: Bswz[(((ntile*4 + kt)*32 + lane)*16) + j]
//   ntile = col>>4, lane = khalf*16 + (col&15), kt = k>>5, khalf = (k>>4)&1,
//   j = k&15.  Each lane's 16 elements are contiguous (32B).
__global__ __launch_bounds__(256)
void k_conv_b(const float* __restrict__ basis, unsigned short* __restrict__ bswz) {
    int t = blockIdx.x * 256 + threadIdx.x;
    if (t >= N_BASES * DIM * DIM) return;
    int b = t >> 14;            // / (128*128)
    int k = (t >> 7) & 127;     // input dim
    int o = t & 127;            // output dim
    int col   = b * DIM + o;
    int ntile = col >> 4;
    int kt    = k >> 5;
    int khalf = (k >> 4) & 1;
    int j     = k & 15;
    int lane  = khalf * 16 + (col & 15);
    bswz[(((ntile * 4 + kt) * 32 + lane) << 4) + j] = f2bf_bits(basis[t]);
}

// --- zero the scatter accumulator and the BN stat buffer ---------------------
__global__ __launch_bounds__(256)
void k_zero(float* __restrict__ agg, float* __restrict__ stats, int total) {
    int i = blockIdx.x * 256 + threadIdx.x;
    if (i < total) agg[i] = 0.f;
    if (i < 2 * DIM) stats[i] = 0.f;
}

// --- WMMA GEMM: hb[50000 x 512] = A[50000 x 128]bf16 * B[128 x 512]bf16 ------
// One wave per 16x64 output strip. grid = (3125, 8), block = 32 (wave32).
__global__ __launch_bounds__(32)
void k_wmma_gemm(const unsigned short* __restrict__ abf,
                 const unsigned short* __restrict__ bswz,
                 float* __restrict__ hb) {
    const int lane   = threadIdx.x & 31;
    const int m0     = blockIdx.x * 16;
    const int l15    = lane & 15;
    const int khalf  = lane >> 4;
    const int rowA   = m0 + l15;
    const unsigned short* arow = abf + (size_t)rowA * DIM;

    v8f acc[4] = {};   // four 16x16 f32 tiles

    #pragma unroll
    for (int kt = 0; kt < 4; ++kt) {          // K = 128 in steps of 32
        // A fragment (contiguous, matches B swizzle): j=0..15 -> K = kt*32+khalf*16+j.
        // Two aligned 16B vector loads.
        v8us alo = *(const v8us*)(arow + kt * 32 + khalf * 16);
        v8us ahi = *(const v8us*)(arow + kt * 32 + khalf * 16 + 8);
        v16us au;
        #pragma unroll
        for (int j = 0; j < 8; ++j) { au[j] = alo[j]; au[j + 8] = ahi[j]; }
        v16bf a = __builtin_bit_cast(v16bf, au);

        #pragma unroll
        for (int t = 0; t < 4; ++t) {
            // B fragment: one contiguous 32B load per lane (pre-swizzled).
            const int ntile = blockIdx.y * 4 + t;
            v16us bu = *(const v16us*)(bswz + ((((ntile * 4 + kt) * 32) + lane) << 4));
            v16bf b = __builtin_bit_cast(v16bf, bu);
            acc[t] = __builtin_amdgcn_wmma_f32_16x16x32_bf16(
                false, a, false, b, (short)0, acc[t], false, false);
        }
    }

    // D layout: VGPR r <-> M = khalf*8 + r, N = lane&15
    const int rbase = khalf * 8;
    #pragma unroll
    for (int t = 0; t < 4; ++t) {
        const int col = blockIdx.y * 64 + t * 16 + l15;
        #pragma unroll
        for (int r = 0; r < 8; ++r)
            hb[(size_t)(m0 + rbase + r) * NCOLS + col] = acc[t][r];
    }
}

// --- per-edge basis mix + scatter-add ----------------------------------------
// 2 edges per 256-thread block; one thread per output channel.
__global__ __launch_bounds__(256)
void k_edge_scatter(const int* __restrict__ src, const int* __restrict__ dst,
                    const int* __restrict__ etype, const float* __restrict__ norm,
                    const float* __restrict__ comp, const float* __restrict__ hb,
                    float* __restrict__ agg) {
    int e = blockIdx.x * 2 + (threadIdx.x >> 7);
    int o = threadIdx.x & 127;
    if (e >= N_EDGES) return;
    int s = src[e], d = dst[e], r = etype[e];
    float nm = norm[e];
    const float* cr = comp + (size_t)r * N_BASES;
    const float* hrow = hb + (size_t)s * NCOLS;
    float m = cr[0] * hrow[o]
            + cr[1] * hrow[DIM + o]
            + cr[2] * hrow[2 * DIM + o]
            + cr[3] * hrow[3 * DIM + o];
    __hip_atomic_fetch_add(agg + (size_t)d * DIM + o, nm * m,
                           __ATOMIC_RELAXED, __HIP_MEMORY_SCOPE_AGENT);
}

// --- fused bias+relu+residual (in place) and BN statistics -------------------
// block = 128 (one thread per channel), 128 rows per block.
__global__ __launch_bounds__(128)
void k_resid_stats(float* __restrict__ hbuf, const float* __restrict__ x,
                   const float* __restrict__ bias, float* __restrict__ stats) {
    int c = threadIdx.x;
    int row0 = blockIdx.x * 128;
    int rowEnd = row0 + 128; if (rowEnd > N_NODES) rowEnd = N_NODES;
    float b = bias[c];
    float sum = 0.f, ss = 0.f;
    for (int row = row0; row < rowEnd; ++row) {
        size_t idx = (size_t)row * DIM + c;
        float v = hbuf[idx] + b;
        v = v > 0.f ? v : 0.f;          // inner relu
        v += x[idx];                    // residual
        hbuf[idx] = v;
        sum += v; ss += v * v;
    }
    __hip_atomic_fetch_add(stats + c,       sum, __ATOMIC_RELAXED, __HIP_MEMORY_SCOPE_AGENT);
    __hip_atomic_fetch_add(stats + DIM + c, ss,  __ATOMIC_RELAXED, __HIP_MEMORY_SCOPE_AGENT);
}

// --- BN scale/shift from accumulated stats (single 128-thread block) ---------
__global__ __launch_bounds__(128)
void k_bn_params(const float* __restrict__ stats, const float* __restrict__ gamma,
                 const float* __restrict__ beta, float* __restrict__ scsh) {
    int c = threadIdx.x;
    const float inv = 1.f / (float)N_NODES;
    float mean = stats[c] * inv;
    float var  = stats[DIM + c] * inv - mean * mean;   // biased variance
    float sc   = gamma[c] * rsqrtf(var + BN_EPS);
    scsh[c]       = sc;
    scsh[DIM + c] = beta[c] - mean * sc;
}

// --- apply BN + outer relu ---------------------------------------------------
__global__ __launch_bounds__(256)
void k_apply(const float* __restrict__ hbuf, const float* __restrict__ scsh,
             float* __restrict__ out, int total) {
    int i = blockIdx.x * 256 + threadIdx.x;
    if (i >= total) return;
    int c = i & 127;
    float v = hbuf[i] * scsh[c] + scsh[DIM + c];
    out[i] = v > 0.f ? v : 0.f;
}

// ---------------------------------------------------------------------------
extern "C" void kernel_launch(void* const* d_in, const int* in_sizes, int n_in,
                              void* d_out, int out_size, void* d_ws, size_t ws_size,
                              hipStream_t stream) {
    const float* node_feats = (const float*)d_in[0];
    const int*   src        = (const int*)  d_in[1];
    const int*   dst        = (const int*)  d_in[2];
    const int*   etype      = (const int*)  d_in[3];
    const float* norm       = (const float*)d_in[4];
    const float* basis      = (const float*)d_in[5];
    const float* comp       = (const float*)d_in[6];
    const float* bias       = (const float*)d_in[7];
    const float* gamma      = (const float*)d_in[8];
    const float* beta       = (const float*)d_in[9];
    (void)in_sizes; (void)n_in; (void)out_size; (void)ws_size;

    // workspace layout (32B aligned blocks)
    char* ws = (char*)d_ws;
    const size_t SZ_ABF  = (size_t)N_NODES * DIM * sizeof(unsigned short);   // 12.8 MB
    const size_t SZ_BBF  = (size_t)DIM * NCOLS * sizeof(unsigned short);     // 128 KB
    const size_t SZ_HB   = (size_t)N_NODES * NCOLS * sizeof(float);          // 102.4 MB
    const size_t SZ_AGG  = (size_t)N_NODES * DIM * sizeof(float);            // 25.6 MB
    unsigned short* abf  = (unsigned short*)(ws);
    unsigned short* bswz = (unsigned short*)(ws + SZ_ABF);
    float* hb            = (float*)(ws + SZ_ABF + SZ_BBF);
    float* agg           = (float*)(ws + SZ_ABF + SZ_BBF + SZ_HB);           // also h (in place)
    float* stats         = (float*)(ws + SZ_ABF + SZ_BBF + SZ_HB + SZ_AGG);  // 256 f32
    float* scsh          = stats + 2 * DIM;                                  // 256 f32
    float* out           = (float*)d_out;

    const int totalND = N_NODES * DIM;      // 6.4M

    k_conv_a<<<(totalND + 255) / 256, 256, 0, stream>>>(node_feats, abf, totalND);
    k_conv_b<<<(N_BASES * DIM * DIM + 255) / 256, 256, 0, stream>>>(basis, bswz);
    k_zero  <<<(totalND + 255) / 256, 256, 0, stream>>>(agg, stats, totalND);

    dim3 gemmGrid(N_NODES / 16, NCOLS / 64);      // 3125 x 8, exact
    k_wmma_gemm<<<gemmGrid, 32, 0, stream>>>(abf, bswz, hb);

    k_edge_scatter<<<(N_EDGES + 1) / 2, 256, 0, stream>>>(src, dst, etype, norm,
                                                          comp, hb, agg);

    k_resid_stats<<<(N_NODES + 127) / 128, 128, 0, stream>>>(agg, node_feats, bias, stats);
    k_bn_params  <<<1, 128, 0, stream>>>(stats, gamma, beta, scsh);
    k_apply      <<<(totalND + 255) / 256, 256, 0, stream>>>(agg, scsh, out, totalND);
}
